// GaussianMixtureModel_batched_67242007986522
// MI455X (gfx1250) — matrix operand order repaired and checked by reading, not compiled, not run on hardware
//
#include <hip/hip_runtime.h>
#include <hip/hip_bf16.h>

typedef __attribute__((ext_vector_type(16))) _Float16 v16h;
typedef __attribute__((ext_vector_type(2)))  _Float16 v2h;
typedef __attribute__((ext_vector_type(8)))  float    v8f;

#define NS   1024   // samples S
#define ND   256    // design points D
#define NK   8      // mixture components K
#define DD   32     // feature dim d
#define STILE 256   // samples per block in main kernel
#define MPB  4      // matrices per block in cholesky kernel

// ---------------------------------------------------------------------------
// Kernel A0: per-D log_softmax(mix) and sum(log(stds)).  One wave per D.
// ---------------------------------------------------------------------------
__global__ __launch_bounds__(32)
void gmm_prep(const float* __restrict__ mix, const float* __restrict__ stds,
              float* __restrict__ logpi, float* __restrict__ sls) {
    int Dc = blockIdx.x;
    int lane = threadIdx.x;

    // sum of log std over d=32
    float lv = logf(stds[Dc * DD + lane]);
    for (int o = 16; o >= 1; o >>= 1) lv += __shfl_xor(lv, o, 32);
    if (lane == 0) sls[Dc] = lv;

    // log_softmax over K=8 (lanes 0..7 hold values)
    float m = (lane < NK) ? mix[Dc * NK + lane] : -1e30f;
    float mx = m;
    for (int o = 16; o >= 1; o >>= 1) mx = fmaxf(mx, __shfl_xor(mx, o, 32));
    float e = (lane < NK) ? expf(m - mx) : 0.0f;
    float ss = e;
    for (int o = 16; o >= 1; o >>= 1) ss += __shfl_xor(ss, o, 32);
    if (lane < NK) logpi[Dc * NK + lane] = m - mx - logf(ss);
}

// ---------------------------------------------------------------------------
// Kernel A: per (D,k): cov = P P^T + eps I, Cholesky, Linv, t = Linv*mu,
// and the per-component constant. One wave per 32x32 matrix, 4 per block.
// ---------------------------------------------------------------------------
__global__ __launch_bounds__(128)
void gmm_chol(const float* __restrict__ covp, const float* __restrict__ meanp,
              const float* __restrict__ logpi,
              _Float16* __restrict__ linv_h, float* __restrict__ tG,
              float* __restrict__ ccst) {
    __shared__ float sA[MPB][DD][DD + 1];  // P, later Linv
    __shared__ float sC[MPB][DD][DD + 1];  // cov, later chol L

    int wv = threadIdx.x >> 5;
    int lane = threadIdx.x & 31;
    int gm = blockIdx.x * MPB + wv;        // global matrix id = Dc*8 + k

    // stage P (coalesced)
    const float* P = covp + (size_t)gm * (DD * DD);
    for (int idx = lane; idx < DD * DD; idx += 32)
        sA[wv][idx >> 5][idx & 31] = P[idx];
    __syncthreads();

    // cov = P P^T + eps I; lane computes its row
    for (int j = 0; j < DD; ++j) {
        float s = 0.0f;
        for (int m = 0; m < DD; ++m) s += sA[wv][lane][m] * sA[wv][j][m];
        sC[wv][lane][j] = s + ((lane == j) ? 1e-5f : 0.0f);
    }
    __syncthreads();

    // in-place right-looking Cholesky (lower)
    float hld = 0.0f;  // half log det, accumulated in lane 0
    for (int j = 0; j < DD; ++j) {
        float ljj = sqrtf(sC[wv][j][j]);
        if (lane == 0) hld += logf(ljj);
        if (lane == j) sC[wv][j][j] = ljj;
        if (lane > j)  sC[wv][lane][j] /= ljj;
        __syncthreads();
        if (lane > j) {
            float lij = sC[wv][lane][j];
            for (int m = j + 1; m <= lane; ++m)
                sC[wv][lane][m] -= lij * sC[wv][m][j];
        }
        __syncthreads();
    }

    // Linv by forward substitution; lane owns column c (independent per lane)
    {
        int c = lane;
        for (int i = 0; i < c; ++i) sA[wv][i][c] = 0.0f;
        sA[wv][c][c] = 1.0f / sC[wv][c][c];
        for (int i = c + 1; i < DD; ++i) {
            float s = 0.0f;
            for (int m = c; m < i; ++m) s += sC[wv][i][m] * sA[wv][m][c];
            sA[wv][i][c] = -s / sC[wv][i][i];
        }
    }
    __syncthreads();

    // t = Linv * mu
    const float* mu = meanp + (size_t)gm * DD;
    float ti = 0.0f;
    for (int j = 0; j < DD; ++j) ti += sA[wv][lane][j] * mu[j];
    tG[gm * DD + lane] = ti;

    // write Linv as f16 (row-major)
    _Float16* Lh = linv_h + (size_t)gm * (DD * DD) + lane * DD;
    for (int j = 0; j < DD; ++j) Lh[j] = (_Float16)sA[wv][lane][j];

    if (lane == 0) {
        const float LOG2PI = 1.8378770664093453f;
        ccst[gm] = -0.5f * (32.0f * LOG2PI) - hld + logpi[gm];
    }
}

// ---------------------------------------------------------------------------
// Kernel B: heavy path. One block = (D, 256-sample tile); 8 waves = 8 comps.
// z = Linv*y - t computed in ONE step per tile: v_wmma_f32_16x16x32_f16 with
// the C operand carrying -t broadcast across columns (C[r] is column-invariant
// in the 16x16 f32 C layout).  maha = sum z^2.
// ---------------------------------------------------------------------------
__global__ __launch_bounds__(256)
void gmm_main(const float* __restrict__ data, const float* __restrict__ means,
              const float* __restrict__ stds,
              const _Float16* __restrict__ linv_h, const float* __restrict__ tG,
              const float* __restrict__ ccst, const float* __restrict__ sls,
              float* __restrict__ out) {
    // y staged in f16 *already in B-fragment layout*: [subtile][lane][16 halves]
    __shared__ __align__(32) _Float16 sYf[(STILE / 16) * 32 * 16];   // 16 KiB
    __shared__ float sP[NK][2][STILE + 16];                          // partial row sums
    __shared__ float sCC[NK];

    const int dIdx = blockIdx.y;
    const int s0   = blockIdx.x * STILE;
    const int tid  = threadIdx.x;
    const int lane = tid & 31;
    const int wv   = tid >> 5;            // component k
    const int lo   = lane & 15;
    const int hi   = lane >> 4;           // 0 or 1

    // ---- stage y = (x - mean)/std into LDS (f16, fragment layout) ----
    {
        int jp   = tid & 15;              // feature pair 0..15 -> j0 = 2*jp
        int j0   = jp << 1;
        int srow = tid >> 4;              // 0..15
        float mn0 = means[dIdx * DD + j0];
        float mn1 = means[dIdx * DD + j0 + 1];
        float is0 = 1.0f / stds[dIdx * DD + j0];
        float is1 = 1.0f / stds[dIdx * DD + j0 + 1];
        for (int i = 0; i < 16; ++i) {
            int sl = srow + (i << 4);     // sample within tile
            const float* p = &data[((size_t)(s0 + sl) * ND + dIdx) * DD + j0];
            float y0 = (p[0] - mn0) * is0;
            float y1 = (p[1] - mn1) * is1;
            // B layout: lane entry = (s%16) + 16*(j>=16); half index = j&15
            int st = sl >> 4;
            int le = (sl & 15) + (j0 & 16);
            v2h pk; pk[0] = (_Float16)y0; pk[1] = (_Float16)y1;
            *(v2h*)&sYf[(st * 32 + le) * 16 + (j0 & 15)] = pk;
        }
    }

    // ---- A fragments: rows 0..15 / 16..31 of this component's Linv ----
    // A k order used here (same as B staging): K = (hi<<4) + h
    const _Float16* Lk = linv_h + (size_t)(dIdx * NK + wv) * (DD * DD);
    v16h a0, a1;
    #pragma unroll
    for (int h = 0; h < 16; ++h) {
        int kA = (hi << 4) + h;
        a0[h] = Lk[lo * DD + kA];
        a1[h] = Lk[(lo + 16) * DD + kA];
    }

    // ---- C bias fragments: C[r] = -t[row], row = r + 8*hi (+16 for a1) ----
    const float* tk = tG + (dIdx * NK + wv) * DD;
    v8f cneg0, cneg1;
    #pragma unroll
    for (int r = 0; r < 8; ++r) {
        cneg0[r] = -tk[r + hi * 8];
        cneg1[r] = -tk[16 + r + hi * 8];
    }

    float cc = ccst[dIdx * NK + wv];
    if (lane == 0) sCC[wv] = cc;

    __syncthreads();

    // ---- 16 subtiles x 2 row-halves ----
    #pragma unroll 4
    for (int st = 0; st < STILE / 16; ++st) {
        v16h b = *(const v16h*)&sYf[(st * 32 + lane) * 16];  // 2x ds_load_b128
        v8f z0 = __builtin_amdgcn_wmma_f32_16x16x32_f16(false, a0, false, b,
                                                        (short)0, cneg0, false, false);
        v8f z1 = __builtin_amdgcn_wmma_f32_16x16x32_f16(false, a1, false, b,
                                                        (short)0, cneg1, false, false);
        float part = 0.0f;
        #pragma unroll
        for (int r = 0; r < 8; ++r) {
            part = fmaf(z0[r], z0[r], part);
            part = fmaf(z1[r], z1[r], part);
        }
        sP[wv][hi][st * 16 + lo] = part;   // both halves store; combined later
    }

    __syncthreads();

    // ---- logsumexp over K per sample (tid == local sample) ----
    {
        float v[NK], mx = -1e30f;
        #pragma unroll
        for (int k = 0; k < NK; ++k) {
            float maha = sP[k][0][tid] + sP[k][1][tid];
            v[k] = sCC[k] - 0.5f * maha;
            mx = fmaxf(mx, v[k]);
        }
        float ss = 0.0f;
        #pragma unroll
        for (int k = 0; k < NK; ++k) ss += expf(v[k] - mx);
        out[(size_t)(s0 + tid) * ND + dIdx] = mx + logf(ss) - sls[dIdx];
    }
}

// ---------------------------------------------------------------------------
extern "C" void kernel_launch(void* const* d_in, const int* in_sizes, int n_in,
                              void* d_out, int out_size, void* d_ws, size_t ws_size,
                              hipStream_t stream) {
    (void)in_sizes; (void)n_in; (void)out_size; (void)ws_size;
    const float* data  = (const float*)d_in[0];   // [S,D,d]
    const float* mix   = (const float*)d_in[1];   // [D,K]
    const float* meanp = (const float*)d_in[2];   // [D,K,d]
    const float* covp  = (const float*)d_in[3];   // [D,K,d,d]
    const float* dsm   = (const float*)d_in[4];   // [D,d]
    const float* dss   = (const float*)d_in[5];   // [D,d]
    float* out = (float*)d_out;                   // [S,D]

    // workspace layout
    char* ws = (char*)d_ws;
    _Float16* linv_h = (_Float16*)ws;                        // D*K*32*32 f16 = 4 MiB
    float* tG    = (float*)(ws + (size_t)ND * NK * DD * DD * 2);
    float* ccst  = tG + (size_t)ND * NK * DD;                // D*K
    float* logpi = ccst + (size_t)ND * NK;                   // D*K
    float* sls   = logpi + (size_t)ND * NK;                  // D

    gmm_prep<<<ND, 32, 0, stream>>>(mix, dss, logpi, sls);
    gmm_chol<<<(ND * NK) / MPB, 128, 0, stream>>>(covp, meanp, logpi,
                                                  linv_h, tG, ccst);
    dim3 grid(NS / STILE, ND);
    gmm_main<<<grid, 256, 0, stream>>>(data, dsm, dss, linv_h, tG, ccst, sls, out);
}
